// MultiHeadLatentAttention_28879360098372
// MI455X (gfx1250) — compile-verified, matrix operand order checked by reading, not run on hardware
//
#include <hip/hip_runtime.h>
#include <hip/hip_bf16.h>

// ---------------------------------------------------------------------------
// MLA prefill for MI455X (gfx1250): bf16 WMMA + TDM (tensor_load_to_lds)
// double-buffered staging, fused flash attention over the latent KV cache.
// ---------------------------------------------------------------------------

typedef __bf16 bf16;
typedef __attribute__((ext_vector_type(8)))  __bf16 bf16x8;
typedef __attribute__((ext_vector_type(16))) __bf16 bf16x16;
typedef __attribute__((ext_vector_type(8)))  float  f32x8;
typedef __attribute__((ext_vector_type(4)))  unsigned int u32x4;
typedef __attribute__((ext_vector_type(8)))  int i32x8;
typedef __attribute__((ext_vector_type(4)))  int i32x4;

#define NB    4
#define SQ    2048
#define BS_   (NB * SQ)        // 8192
#define DIM_  1024
#define NH_   12
#define NOPE_ 128
#define ROPE_ 64
#define QKH_  192
#define VH_   128
#define KVR_  512
#define QC    576              // KVR + ROPE  (concat dim for scores)
#define SM_SCALE_ 0.07216878364870323f   // 1/sqrt(192)

// ---------------------------------------------------------------------------
// WMMA helpers (CDNA5 16x16x32 bf16, f32 accum)
// ---------------------------------------------------------------------------
__device__ __forceinline__ f32x8 wmma_bf16(bf16x16 a, bf16x16 b, f32x8 c) {
  return __builtin_amdgcn_wmma_f32_16x16x32_bf16(false, a, false, b, (short)0, c,
                                                 false, false);
}

// 16x32 bf16 operand fragment (A-style; B operand identical when stored NxK
// row-major, K contiguous).  ISA 7.12.2: two 16-byte chunks per lane.
__device__ __forceinline__ bf16x16 load_frag(const bf16* base, int ld, int lane) {
  const int r  = lane & 15;
  const int kg = (lane >> 4) << 3;          // 0 or 8
  const bf16* p = base + (size_t)r * ld;
  bf16x8 lo = *(const bf16x8*)(p + kg);
  bf16x8 hi = *(const bf16x8*)(p + 16 + kg);
  return __builtin_shufflevector(lo, hi, 0, 1, 2, 3, 4, 5, 6, 7,
                                         8, 9, 10, 11, 12, 13, 14, 15);
}

__device__ __forceinline__ float rowred_max(float x) {
#pragma unroll
  for (int off = 1; off < 16; off <<= 1) x = fmaxf(x, __shfl_xor(x, off, 16));
  return x;
}
__device__ __forceinline__ float rowred_sum(float x) {
#pragma unroll
  for (int off = 1; off < 16; off <<= 1) x += __shfl_xor(x, off, 16);
  return x;
}

// ---------------------------------------------------------------------------
// TDM: stage a 2-D tile (tile_rows x tile_cols bf16, row stride = ld elements)
// from global into packed LDS.  Rows/cols beyond (rows_rem, cols_rem) are
// zero-filled by the TDM (OOB reads return zero).  D# packing per ISA 8.3/8.4.
// All inputs must be workgroup-uniform (descriptor lives in SGPRs).
// ---------------------------------------------------------------------------
__device__ __forceinline__ void tdm_load_2d(unsigned lds_off_bytes,
                                            const bf16* gptr,
                                            unsigned rows_rem, unsigned cols_rem,
                                            unsigned tile_rows, unsigned tile_cols,
                                            unsigned ld_elems) {
  unsigned long long ga = (unsigned long long)(uintptr_t)gptr;
  u32x4 g0;
  g0[0] = 1u;                                      // count=1, user descriptor
  g0[1] = lds_off_bytes;                           // lds_addr
  g0[2] = (unsigned)(ga & 0xffffffffu);            // global_addr[31:0]
  g0[3] = (unsigned)((ga >> 32) & 0x01ffffffu)     // global_addr[56:32]
          | (2u << 30);                            // type = 2 ("image")
  unsigned long long st = (unsigned long long)ld_elems;
  i32x8 g1;
  g1[0] = (int)(1u << 16);                         // wg_mask=0, data_size=1 (2B)
  g1[1] = (int)((cols_rem & 0xffffu) << 16);       // tensor_dim0[15:0]
  g1[2] = (int)((cols_rem >> 16) | ((rows_rem & 0xffffu) << 16)); // td0 hi, td1 lo
  g1[3] = (int)((rows_rem >> 16) | (tile_cols << 16));            // td1 hi, tile_dim0
  g1[4] = (int)(tile_rows & 0xffffu);              // tile_dim1 (tile_dim2 = 0)
  g1[5] = (int)(st & 0xffffffffu);                 // tensor_dim0_stride lo
  g1[6] = (int)((st >> 32) & 0xffffu);             // stride hi (dim1_stride = 0)
  g1[7] = 0;
  i32x4 z4 = {0, 0, 0, 0};
  i32x8 z8 = {0, 0, 0, 0, 0, 0, 0, 0};
  __builtin_amdgcn_tensor_load_to_lds(g0, g1, z4, z4, z8, 0);
}

__device__ __forceinline__ unsigned lds_off(const void* p) {
  return (unsigned)(uintptr_t)p;                   // low 32 bits = LDS byte offset
}

// ---------------------------------------------------------------------------
// Generic bf16 GEMM:  C[z](M,N) = A[z](M,K) @ B[z](N,K)^T
// Block 128x128, K-step 32, 8 waves (2x4), TDM double-buffered staging.
// ---------------------------------------------------------------------------
__global__ __launch_bounds__(256) void gemm_bf16_k(
    const bf16* __restrict__ A, const bf16* __restrict__ B,
    float* __restrict__ Cf, bf16* __restrict__ Cb,
    int M, int N, int K, int lda, int ldb, int ldc,
    long long sA, long long sB, long long sC)
{
  __shared__ bf16 sAe[2][128 * 32];
  __shared__ bf16 sBe[2][128 * 32];

  const int z = blockIdx.z;
  A += (size_t)z * sA;
  B += (size_t)z * sB;

  const int tid  = threadIdx.x;
  const int lane = tid & 31;
  const int wave = tid >> 5;
  const int wm   = wave >> 2;               // 0..1
  const int wn   = wave & 3;                // 0..3
  const int bm   = blockIdx.y * 128;
  const int bn   = blockIdx.x * 128;

  f32x8 acc[4][2] = {};

  const int nk = K >> 5;                    // K is a multiple of 32 here
  if (wave == 0) {
    tdm_load_2d(lds_off(&sAe[0][0]), A + (size_t)bm * lda,
                (unsigned)(M - bm), (unsigned)K, 128, 32, (unsigned)lda);
    tdm_load_2d(lds_off(&sBe[0][0]), B + (size_t)bn * ldb,
                (unsigned)(N - bn), (unsigned)K, 128, 32, (unsigned)ldb);
  }

  for (int i = 0; i < nk; ++i) {
    const int cur = i & 1;
    const int nxt = cur ^ 1;
    if (wave == 0) {
      if (i + 1 < nk) {
        const int k0n = (i + 1) << 5;
        tdm_load_2d(lds_off(&sAe[nxt][0]), A + (size_t)bm * lda + k0n,
                    (unsigned)(M - bm), (unsigned)(K - k0n), 128, 32, (unsigned)lda);
        tdm_load_2d(lds_off(&sBe[nxt][0]), B + (size_t)bn * ldb + k0n,
                    (unsigned)(N - bn), (unsigned)(K - k0n), 128, 32, (unsigned)ldb);
        __builtin_amdgcn_s_wait_tensorcnt(2);   // oldest A/B pair complete
      } else {
        __builtin_amdgcn_s_wait_tensorcnt(0);
      }
    }
    __syncthreads();

    bf16x16 bf0 = load_frag(&sBe[cur][(wn * 32 + 0)  * 32], 32, lane);
    bf16x16 bf1 = load_frag(&sBe[cur][(wn * 32 + 16) * 32], 32, lane);
#pragma unroll
    for (int mi = 0; mi < 4; ++mi) {
      bf16x16 af = load_frag(&sAe[cur][(wm * 64 + mi * 16) * 32], 32, lane);
      acc[mi][0] = wmma_bf16(af, bf0, acc[mi][0]);
      acc[mi][1] = wmma_bf16(af, bf1, acc[mi][1]);
    }
    __syncthreads();                         // readers done before next overwrite
  }

  const int mh = (lane >> 4) * 8;
  const int nl = lane & 15;
#pragma unroll
  for (int mi = 0; mi < 4; ++mi)
#pragma unroll
    for (int ni = 0; ni < 2; ++ni)
#pragma unroll
      for (int v = 0; v < 8; ++v) {
        int gr = bm + wm * 64 + mi * 16 + mh + v;
        int gc = bn + wn * 32 + ni * 16 + nl;
        if (gr < M && gc < N) {
          float x = acc[mi][ni][v];
          size_t ci = (size_t)z * sC + (size_t)gr * ldc + gc;
          if (Cf) Cf[ci] = x;
          else    Cb[ci] = (bf16)x;
        }
      }
}

// ---------------------------------------------------------------------------
// Fused flash attention over the latent cache.
//   qcat (NH, BS, 576) bf16, kcat (BS, 576) bf16 -> olat (NH, BS, 512) bf16
// Block = 64 q-rows of one (b,h); 8 waves; wave pair owns 16 rows, waves in a
// pair split the 512 c-dim of O.  K tile staged by TDM (double-buffered).
// ---------------------------------------------------------------------------
__global__ __launch_bounds__(256) void mla_attn_k(
    const bf16* __restrict__ qcat, const bf16* __restrict__ kcat,
    bf16* __restrict__ olat)
{
  extern __shared__ char smem_raw[];
  bf16* lds_k[2];
  lds_k[0] = (bf16*)smem_raw;                                   // [32][576]
  lds_k[1] = (bf16*)(smem_raw + 32 * QC * 2);                   // [32][576]
  bf16* lds_vt = (bf16*)(smem_raw + 2 * 32 * QC * 2);           // [512][32]
  bf16* lds_p  = (bf16*)(smem_raw + 2 * 32 * QC * 2 + 512 * 32 * 2); // [8][16][32]

  const int tid    = threadIdx.x;
  const int lane   = tid & 31;
  const int wave   = tid >> 5;
  const int rowgrp = wave >> 1;           // 0..3
  const int chalf  = wave & 1;            // 0..1
  const int b      = blockIdx.z;
  const int h      = blockIdx.y;
  const int qtile0 = blockIdx.x * 64;
  const int qrow0  = qtile0 + rowgrp * 16;

  const int mh = (lane >> 4) * 8;
  const int nl = lane & 15;
  const int c0 = chalf * 256;

  // Q fragments for this wave's 16 rows stay resident (18 k-tiles of 32)
  const size_t qbase = ((size_t)h * BS_ + (size_t)b * SQ + qrow0) * QC;
  bf16x16 qf[18];
#pragma unroll
  for (int kt = 0; kt < 18; ++kt)
    qf[kt] = load_frag(qcat + qbase + kt * 32, QC, lane);

  f32x8 acc[16] = {};
  float rowmax[8], rowsum[8];
#pragma unroll
  for (int v = 0; v < 8; ++v) { rowmax[v] = -INFINITY; rowsum[v] = 0.f; }

  const size_t kb = (size_t)b * SQ;
  const int nIter = (qtile0 + 64) >> 5;     // key tiles of 32

  if (wave == 0)
    tdm_load_2d(lds_off(lds_k[0]), kcat + kb * QC, 32, QC, 32, QC, QC);

  for (int i = 0; i < nIter; ++i) {
    const int t0  = i << 5;
    const int cur = i & 1;
    const int nxt = cur ^ 1;
    if (wave == 0) {
      if (i + 1 < nIter) {
        tdm_load_2d(lds_off(lds_k[nxt]), kcat + (kb + t0 + 32) * QC,
                    32, QC, 32, QC, QC);
        __builtin_amdgcn_s_wait_tensorcnt(1);
      } else {
        __builtin_amdgcn_s_wait_tensorcnt(0);
      }
    }
    __syncthreads();

    // build V^T (c-major) from the staged K tile
    const bf16* kt_lds = lds_k[cur];
    for (int idx = tid; idx < 32 * 64; idx += 256) {
      int row = idx >> 6;              // 0..31
      int c8  = (idx & 63) * 8;        // 0..504
      bf16x8 d = *(const bf16x8*)(kt_lds + row * QC + c8);
#pragma unroll
      for (int j = 0; j < 8; ++j) lds_vt[(c8 + j) * 32 + row] = d[j];
    }
    __syncthreads();

    // scores: 16 x 32, K = 576 (both waves of the pair compute them)
    f32x8 s0 = {}, s1 = {};
#pragma unroll
    for (int kt = 0; kt < 18; ++kt) {
      bf16x16 k0 = load_frag(kt_lds + 0  * QC + kt * 32, QC, lane);
      bf16x16 k1 = load_frag(kt_lds + 16 * QC + kt * 32, QC, lane);
      s0 = wmma_bf16(qf[kt], k0, s0);
      s1 = wmma_bf16(qf[kt], k1, s1);
    }

    // online softmax (per C-tile row v), causal mask applied analytically
    bf16* pw = lds_p + wave * (16 * 32);
#pragma unroll
    for (int v = 0; v < 8; ++v) {
      int   qi = qrow0 + mh + v;
      float a0 = (t0 + nl      <= qi) ? s0[v] * SM_SCALE_ : -INFINITY;
      float a1 = (t0 + 16 + nl <= qi) ? s1[v] * SM_SCALE_ : -INFINITY;
      float tm = rowred_max(fmaxf(a0, a1));
      float nm = fmaxf(rowmax[v], tm);
      float ok = (nm == -INFINITY) ? 0.f : 1.f;
      float sc = ok * expf(rowmax[v] - nm);
      float p0 = ok * expf(a0 - nm);
      float p1 = ok * expf(a1 - nm);
      rowsum[v] = rowsum[v] * sc + rowred_sum(p0 + p1);
      rowmax[v] = nm;
#pragma unroll
      for (int n = 0; n < 16; ++n) acc[n][v] *= sc;
      pw[(mh + v) * 32 + nl]      = (bf16)p0;
      pw[(mh + v) * 32 + 16 + nl] = (bf16)p1;
    }

    // O += P (16x32) @ V (32 x 256-half); DS ops are in-order per wave
    bf16x16 pf = load_frag(pw, 32, lane);
#pragma unroll
    for (int n = 0; n < 16; ++n) {
      bf16x16 vf = load_frag(lds_vt + (c0 + n * 16) * 32, 32, lane);
      acc[n] = wmma_bf16(pf, vf, acc[n]);
    }
    __syncthreads();                  // readers done before TDM overwrites cur
  }

#pragma unroll
  for (int v = 0; v < 8; ++v) {
    float inv = 1.f / rowsum[v];
    size_t orow = ((size_t)h * BS_ + (size_t)b * SQ + qrow0 + mh + v) * (size_t)KVR_ + c0;
#pragma unroll
    for (int n = 0; n < 16; ++n)
      olat[orow + n * 16 + nl] = (bf16)(acc[n][v] * inv);
  }
}

// ---------------------------------------------------------------------------
// Elementwise / prep kernels
// ---------------------------------------------------------------------------
__global__ void cast_bf16_k(const float* __restrict__ s, bf16* __restrict__ d,
                            long long n) {
  long long i = (long long)blockIdx.x * blockDim.x + threadIdx.x;
  long long stride = (long long)gridDim.x * blockDim.x;
  for (; i < n; i += stride) d[i] = (bf16)s[i];
}

// wbT[h][c][d] = wkv_b[(h*256 + d)*512 + c], d < 128 (NOPE rows, transposed)
__global__ void build_wbT_k(const float* __restrict__ src, bf16* __restrict__ dst) {
  long long n = (long long)NH_ * KVR_ * NOPE_;
  long long i = (long long)blockIdx.x * blockDim.x + threadIdx.x;
  long long stride = (long long)gridDim.x * blockDim.x;
  for (; i < n; i += stride) {
    int d = (int)(i & (NOPE_ - 1));
    long long t = i >> 7;
    int c = (int)(t & (KVR_ - 1));
    int h = (int)(t >> 9);
    dst[i] = (bf16)src[((size_t)(h * 256 + d)) * KVR_ + c];
  }
}

// qnope[h][bs][d] = q_f32[bs][h*192 + d]
__global__ void prep_qnope_k(const float* __restrict__ q, bf16* __restrict__ dst) {
  long long n = (long long)NH_ * BS_ * NOPE_;
  long long i = (long long)blockIdx.x * blockDim.x + threadIdx.x;
  long long stride = (long long)gridDim.x * blockDim.x;
  for (; i < n; i += stride) {
    int d = (int)(i & (NOPE_ - 1));
    long long t = i >> 7;
    int bs = (int)(t % BS_);
    int h  = (int)(t / BS_);
    dst[i] = (bf16)q[(size_t)bs * (NH_ * QKH_) + h * QKH_ + d];
  }
}

// RoPE on q_pe -> qcat[h][bs][512 + 2j .. 2j+1]
__global__ void prep_qpe_k(const float* __restrict__ q, const float* __restrict__ fc,
                           bf16* __restrict__ qcat) {
  long long n = (long long)NH_ * BS_ * (ROPE_ / 2);
  long long i = (long long)blockIdx.x * blockDim.x + threadIdx.x;
  long long stride = (long long)gridDim.x * blockDim.x;
  for (; i < n; i += stride) {
    int j = (int)(i & 31);
    long long t = i >> 5;
    int bs = (int)(t % BS_);
    int h  = (int)(t / BS_);
    int s  = bs & (SQ - 1);
    const float* qr = q + (size_t)bs * (NH_ * QKH_) + h * QKH_ + NOPE_;
    float xr = qr[2 * j], xi = qr[2 * j + 1];
    float cc = fc[(size_t)s * ROPE_ + 2 * j];
    float sn = fc[(size_t)s * ROPE_ + 2 * j + 1];
    bf16* o = qcat + ((size_t)h * BS_ + bs) * QC + KVR_;
    o[2 * j]     = (bf16)(xr * cc - xi * sn);
    o[2 * j + 1] = (bf16)(xr * sn + xi * cc);
  }
}

// rmsnorm(kv) + RoPE(k_pe) -> kcat (BS, 576)
__global__ __launch_bounds__(256) void prep_kv_k(
    const float* __restrict__ kvf, const float* __restrict__ w,
    const float* __restrict__ fc, bf16* __restrict__ kcat) {
  int bs = blockIdx.x;
  int tid = threadIdx.x;
  const float* row = kvf + (size_t)bs * QC;
  __shared__ float red[256];
  float ss = 0.f;
  for (int c = tid; c < KVR_; c += 256) { float x = row[c]; ss += x * x; }
  red[tid] = ss;
  __syncthreads();
  for (int o = 128; o > 0; o >>= 1) {
    if (tid < o) red[tid] += red[tid + o];
    __syncthreads();
  }
  float rms = rsqrtf(red[0] / (float)KVR_ + 1e-6f);
  bf16* out = kcat + (size_t)bs * QC;
  for (int c = tid; c < KVR_; c += 256) out[c] = (bf16)(row[c] * rms * w[c]);
  if (tid < ROPE_ / 2) {
    int s = bs & (SQ - 1), j = tid;
    float xr = row[KVR_ + 2 * j], xi = row[KVR_ + 2 * j + 1];
    float cc = fc[(size_t)s * ROPE_ + 2 * j];
    float sn = fc[(size_t)s * ROPE_ + 2 * j + 1];
    out[KVR_ + 2 * j]     = (bf16)(xr * cc - xi * sn);
    out[KVR_ + 2 * j + 1] = (bf16)(xr * sn + xi * cc);
  }
}

// ---------------------------------------------------------------------------
// Host
// ---------------------------------------------------------------------------
static void launch_gemm(hipStream_t st, const bf16* A, const bf16* B,
                        float* Cf, bf16* Cb, int M, int N, int K,
                        int lda, int ldb, int ldc,
                        long long sA, long long sB, long long sC, int nz) {
  dim3 g((N + 127) / 128, (M + 127) / 128, nz);
  gemm_bf16_k<<<g, 256, 0, st>>>(A, B, Cf, Cb, M, N, K, lda, ldb, ldc, sA, sB, sC);
}

extern "C" void kernel_launch(void* const* d_in, const int* in_sizes, int n_in,
                              void* d_out, int out_size, void* d_ws, size_t ws_size,
                              hipStream_t stream) {
  const float* x    = (const float*)d_in[0];
  const float* wq   = (const float*)d_in[1];
  const float* wkva = (const float*)d_in[2];
  const float* kvw  = (const float*)d_in[3];
  const float* wkvb = (const float*)d_in[4];
  const float* wo   = (const float*)d_in[5];
  const float* fc   = (const float*)d_in[6];
  // d_in[7] = mask (unused: causal applied analytically), d_in[8] = start_pos (0)
  float* out = (float*)d_out;

  char* ws = (char*)d_ws;
  size_t off = 0;
  auto take = [&](size_t bytes) -> char* {
    char* p = ws + off;
    off += (bytes + 255) & ~(size_t)255;
    return p;
  };

  bf16*  x_bf    = (bf16*)take((size_t)BS_ * DIM_ * 2);
  bf16*  wq_bf   = (bf16*)take((size_t)NH_ * QKH_ * DIM_ * 2);
  bf16*  wkva_bf = (bf16*)take((size_t)QC * DIM_ * 2);
  bf16*  wkvb_bf = (bf16*)take((size_t)NH_ * 256 * KVR_ * 2);
  bf16*  wo_bf   = (bf16*)take((size_t)DIM_ * NH_ * VH_ * 2);
  bf16*  wbT     = (bf16*)take((size_t)NH_ * KVR_ * NOPE_ * 2);
  float* q_f32   = (float*)take((size_t)BS_ * NH_ * QKH_ * 4);
  float* kv_f32  = (float*)take((size_t)BS_ * QC * 4);
  bf16*  qnope   = (bf16*)take((size_t)NH_ * BS_ * NOPE_ * 2);
  bf16*  qcat    = (bf16*)take((size_t)NH_ * BS_ * QC * 2);
  bf16*  kcat    = (bf16*)take((size_t)BS_ * QC * 2);
  bf16*  olat    = (bf16*)take((size_t)NH_ * BS_ * KVR_ * 2);
  bf16*  oc      = (bf16*)take((size_t)BS_ * NH_ * VH_ * 2);

  // 1) precision conversion (activations + weights -> bf16)
  cast_bf16_k<<<1024, 256, 0, stream>>>(x,    x_bf,    (long long)BS_ * DIM_);
  cast_bf16_k<<<512,  256, 0, stream>>>(wq,   wq_bf,   (long long)NH_ * QKH_ * DIM_);
  cast_bf16_k<<<256,  256, 0, stream>>>(wkva, wkva_bf, (long long)QC * DIM_);
  cast_bf16_k<<<512,  256, 0, stream>>>(wkvb, wkvb_bf, (long long)NH_ * 256 * KVR_);
  cast_bf16_k<<<512,  256, 0, stream>>>(wo,   wo_bf,   (long long)DIM_ * NH_ * VH_);
  build_wbT_k<<<512, 256, 0, stream>>>(wkvb, wbT);

  // 2) projections: q = x @ wq^T ; kv_full = x @ wkv_a^T
  launch_gemm(stream, x_bf, wq_bf,   q_f32,  nullptr, BS_, NH_ * QKH_, DIM_,
              DIM_, DIM_, NH_ * QKH_, 0, 0, 0, 1);
  launch_gemm(stream, x_bf, wkva_bf, kv_f32, nullptr, BS_, QC, DIM_,
              DIM_, DIM_, QC, 0, 0, 0, 1);

  // 3) prep: split q, RoPE(q_pe), rmsnorm(kv)+RoPE(k_pe)
  prep_qnope_k<<<1024, 256, 0, stream>>>(q_f32, qnope);
  prep_qpe_k<<<512, 256, 0, stream>>>(q_f32, fc, qcat);
  prep_kv_k<<<BS_, 256, 0, stream>>>(kv_f32, kvw, fc, kcat);

  // 4) absorbed q: qcat[h][:, :512] = qnope_h @ wbT_h^T   (batched over heads)
  launch_gemm(stream, qnope, wbT, nullptr, qcat, BS_, KVR_, NOPE_,
              NOPE_, NOPE_, QC,
              (long long)BS_ * NOPE_, (long long)KVR_ * NOPE_,
              (long long)BS_ * QC, NH_);

  // 5) fused flash attention over latent cache (TDM-staged K tiles)
  {
    size_t smem = (size_t)2 * 32 * QC * 2 + (size_t)512 * 32 * 2
                + (size_t)8 * 16 * 32 * 2;
    dim3 g(SQ / 64, NH_, NB);
    mla_attn_k<<<g, 256, smem, stream>>>(qcat, kcat, olat);
  }

  // 6) un-absorb V: oc[:, h*128:(h+1)*128] = olat_h @ Wv_h^T (batched over heads)
  launch_gemm(stream, olat, wkvb_bf + (size_t)NOPE_ * KVR_, nullptr, oc,
              BS_, VH_, KVR_, KVR_, KVR_, NH_ * VH_,
              (long long)BS_ * KVR_, (long long)256 * KVR_, (long long)VH_, NH_);

  // 7) output projection: out = oc @ wo^T  (f32)
  launch_gemm(stream, oc, wo_bf, out, nullptr, BS_, DIM_, NH_ * VH_,
              NH_ * VH_, NH_ * VH_, DIM_, 0, 0, 0, 1);
}
